// QKNormAttention_56616258896075
// MI455X (gfx1250) — matrix operand order repaired and checked by reading, not compiled, not run on hardware
//
#include <hip/hip_runtime.h>

typedef __attribute__((ext_vector_type(16))) _Float16 v16h;
typedef __attribute__((ext_vector_type(8)))  _Float16 h8;
typedef __attribute__((ext_vector_type(8)))  float    v8f;

#define B_    4
#define S_    2048
#define DIM_  2048
#define H_    16
#define HD_   128
#define MROWS (B_*S_)   // 8192

// ---------------------------------------------------------------------------
// CDNA5 async global->LDS copy (ASYNCcnt-tracked) via inline asm.
// ISA (CDNA5 ISA, opcode 98): VDST = LDS address VGPR,
// VADDR = 64-bit global address, SADDR = off. Low 32 bits of a generic
// pointer into LDS are the LDS byte address (flat-aperture truncation rule).
// ---------------------------------------------------------------------------
static __device__ __forceinline__ void async_copy16(const _Float16* g, _Float16* l) {
  const unsigned lds = (unsigned)(unsigned long long)l;
  const unsigned long long ga = (unsigned long long)g;
  asm volatile("global_load_async_to_lds_b128 %0, %1, off"
               :: "v"(lds), "v"(ga) : "memory");
}
static __device__ __forceinline__ void async_wait0() {
  asm volatile("s_wait_asynccnt 0x0" ::: "memory");
}

// ---------------------------------------------------------------------------
// CDNA5 WMMA VGPR layouts (CDNA5 ISA):
//   A (16x32 f16): lane 0-15 -> M=lane; holds K {g*8..g*8+7, 16+g*8..+7}, g=lane>>4
//   B (32x16 f16): lane 0-15 -> N=lane; holds K {g*8..g*8+7, 16+g*8..+7}, g=lane>>4
//   C/D (16x16 f32): VGPR r, lane l -> M = r + 8*(l>>4), N = l&15
// A-frag loads use 16-byte h8 accesses; B-frag loads gather 16 strided halves.
// ---------------------------------------------------------------------------
static __device__ __forceinline__ v16h combine(h8 lo, h8 hi) {
  union { struct { h8 lo, hi; } p; v16h v; } u;
  u.p.lo = lo; u.p.hi = hi;
  return u.v;
}

static __device__ __forceinline__ v16h load_a_frag(const _Float16* base, int ld, int kOff) {
  const int lane = threadIdx.x & 31;
  const int g = lane >> 4;
  const _Float16* r = base + (lane & 15) * ld + kOff;
  h8 lo = *(const h8*)(r + g * 8);
  h8 hi = *(const h8*)(r + 16 + g * 8);
  return combine(lo, hi);
}

// B-frag from a tile stored [k][n] row-major: n = lane&15; elements hold K {g*8..g*8+7, 16+g*8..+7}.
static __device__ __forceinline__ v16h load_b_frag_kn(const _Float16* base, int ld) {
  const int lane = threadIdx.x & 31;
  const int g = lane >> 4;
  const _Float16* r = base + (lane & 15);
  v16h v;
#pragma unroll
  for (int s = 0; s < 8; ++s) {
    v[s] = r[(g * 8 + s) * ld];
    v[8 + s] = r[(16 + g * 8 + s) * ld];
  }
  return v;
}

// ---------------------------------------------------------------------------
__global__ void cast_f32_to_f16(const float* __restrict__ src,
                                _Float16* __restrict__ dst, int n8) {
  for (int i = blockIdx.x * blockDim.x + threadIdx.x; i < n8;
       i += gridDim.x * blockDim.x) {
    const float4 a = *(const float4*)(src + (size_t)i * 8);
    const float4 b = *(const float4*)(src + (size_t)i * 8 + 4);
    h8 o;
    o[0] = (_Float16)a.x; o[1] = (_Float16)a.y; o[2] = (_Float16)a.z; o[3] = (_Float16)a.w;
    o[4] = (_Float16)b.x; o[5] = (_Float16)b.y; o[6] = (_Float16)b.z; o[7] = (_Float16)b.w;
    *(h8*)(dst + (size_t)i * 8) = o;
  }
}

// ---------------------------------------------------------------------------
// C[M,N] = A[M,K] * W[N,K]^T  (matches x @ w.T). Block tile 128x128, BK=32,
// 256 threads = 8 waves (4x2), wave tile 32x64 = 2x4 WMMA accumulators.
// Double-buffered LDS. A tile streamed with async global->LDS copies;
// W tile staged through registers and stored transposed [k][n]; B-frags
// gather 16 strided halves per lane (k stride 136).
// mode 0: fp32 row-major to Cf.   mode 1: f16 head-transposed to Ch (B,H,S,HD).
// ---------------------------------------------------------------------------
__global__ __launch_bounds__(256)
void gemm_nt_f16(const _Float16* __restrict__ A, const _Float16* __restrict__ W,
                 float* __restrict__ Cf, _Float16* __restrict__ Ch,
                 int M, int N, int K, int mode) {
  __shared__ _Float16 As[2][128][40];    // [m][k]
  __shared__ _Float16 Bt[2][32][136];    // [k][n]  (transposed W tile)
  const int tid  = threadIdx.x;
  const int lane = tid & 31;
  const int wave = tid >> 5;
  const int g = lane >> 4, ln = lane & 15;
  const int wm = (wave >> 1) * 32;
  const int wn = (wave & 1) * 64;
  const int mB = blockIdx.y * 128;
  const int nB = blockIdx.x * 128;

  // per-thread staging: chunks c and c+256 of 512 (128 rows x 4 h8-chunks)
  const int r0 = tid >> 2,         c0 = (tid & 3) << 3;
  const int r1 = (tid + 256) >> 2, c1 = c0;           // (tid+256)&3 == tid&3
  h8 wr[2];

  v8f acc[2][4] = {};
  const int nk = K >> 5;

  // prologue: stage slab 0
  async_copy16(A + (size_t)(mB + r0) * K + c0, &As[0][r0][c0]);
  async_copy16(A + (size_t)(mB + r1) * K + c1, &As[0][r1][c1]);
  wr[0] = *(const h8*)(W + (size_t)(nB + r0) * K + c0);
  wr[1] = *(const h8*)(W + (size_t)(nB + r1) * K + c1);
#pragma unroll
  for (int i = 0; i < 8; ++i) Bt[0][c0 + i][r0] = wr[0][i];
#pragma unroll
  for (int i = 0; i < 8; ++i) Bt[0][c1 + i][r1] = wr[1][i];

  for (int kt = 0; kt < nk; ++kt) {
    const int cur = kt & 1;
    async_wait0();       // this wave's copies into buf[cur] are done
    __syncthreads();     // everyone's copies + transposed stores visible
    if (kt + 1 < nk) {   // stage next slab while current WMMAs run
      const int nxt = (kt + 1) & 1;
      const size_t ko = (size_t)(kt + 1) * 32;
      async_copy16(A + (size_t)(mB + r0) * K + ko + c0, &As[nxt][r0][c0]);
      async_copy16(A + (size_t)(mB + r1) * K + ko + c1, &As[nxt][r1][c1]);
      wr[0] = *(const h8*)(W + (size_t)(nB + r0) * K + ko + c0);
      wr[1] = *(const h8*)(W + (size_t)(nB + r1) * K + ko + c1);
    }
    if (kt + 2 < nk)     // keep the slab after next flowing into cache
      __builtin_prefetch(A + (size_t)(mB + r0) * K + (size_t)(kt + 2) * 32 + c0, 0, 1);

    v16h af[2], bf[4];
#pragma unroll
    for (int i = 0; i < 2; ++i) af[i] = load_a_frag(&As[cur][wm + i * 16][0], 40, 0);
#pragma unroll
    for (int j = 0; j < 4; ++j) bf[j] = load_b_frag_kn(&Bt[cur][0][wn + j * 16], 136);
#pragma unroll
    for (int i = 0; i < 2; ++i) {
#pragma unroll
      for (int j = 0; j < 4; ++j) {
        acc[i][j] = __builtin_amdgcn_wmma_f32_16x16x32_f16(
            false, af[i], false, bf[j], (short)0, acc[i][j], false, false);
      }
    }

    if (kt + 1 < nk) {   // commit transposed W slab to the other buffer
      const int nxt = (kt + 1) & 1;
#pragma unroll
      for (int i = 0; i < 8; ++i) Bt[nxt][c0 + i][r0] = wr[0][i];
#pragma unroll
      for (int i = 0; i < 8; ++i) Bt[nxt][c1 + i][r1] = wr[1][i];
    }
  }

#pragma unroll
  for (int i = 0; i < 2; ++i) {
#pragma unroll
    for (int j = 0; j < 4; ++j) {
#pragma unroll
      for (int r = 0; r < 8; ++r) {
        const int m = mB + wm + i * 16 + r + 8 * g;
        const int n = nB + wn + j * 16 + ln;
        const float v = acc[i][j][r];
        if (mode == 0) {
          Cf[(size_t)m * N + n] = v;
        } else {
          const int b = m >> 11, s = m & (S_ - 1);     // m = b*S + s
          const int h = n >> 7,  d = n & (HD_ - 1);    // n = h*HD + d
          Ch[(((size_t)b * H_ + h) * S_ + s) * HD_ + d] = (_Float16)v;
        }
      }
    }
  }
}

// ---------------------------------------------------------------------------
// Per-(b,h,s) head RMSNorm + rotate-half RoPE, in place on f16 (B,H,S,HD).
// ---------------------------------------------------------------------------
__global__ __launch_bounds__(128)
void rmsnorm_rope(_Float16* __restrict__ qb, _Float16* __restrict__ kb,
                  const float* __restrict__ qw, const float* __restrict__ kw,
                  const float* __restrict__ cosT, const float* __restrict__ sinT) {
  _Float16* buf  = (blockIdx.y == 0) ? qb : kb;
  const float* w = (blockIdx.y == 0) ? qw : kw;
  const int row = blockIdx.x;
  const int s = row & (S_ - 1);
  const int d = threadIdx.x;
  _Float16* p = buf + (size_t)row * HD_;
  const float x = (float)p[d];

  __shared__ float red[128];
  __shared__ float tmp[128];
  red[d] = x * x;
  __syncthreads();
  for (int off = 64; off > 0; off >>= 1) {
    if (d < off) red[d] += red[d + off];
    __syncthreads();
  }
  const float rms = rsqrtf(red[0] * (1.0f / HD_) + 1e-6f);
  const float nx = x * rms * w[d];
  tmp[d] = nx;
  __syncthreads();
  const float rh  = (d < 64) ? -tmp[d + 64] : tmp[d - 64];
  const float out = nx * cosT[s * HD_ + d] + rh * sinT[s * HD_ + d];
  p[d] = (_Float16)out;
}

// ---------------------------------------------------------------------------
// Flash attention per (b,h). 256 threads = 8 waves; each wave owns 16 query
// rows (block covers 128). K stored transposed in LDS ([d][j]); the score
// B-frags gather 16 strided halves per lane; V streamed with async global->LDS copies into [j][d].
// Double-buffered KV; online softmax; analytic causal mask.
// ---------------------------------------------------------------------------
__global__ __launch_bounds__(256)
void flash_attn(const _Float16* __restrict__ qbuf, const _Float16* __restrict__ kbuf,
                const _Float16* __restrict__ vbuf, _Float16* __restrict__ attn) {
  __shared__ _Float16 Kt[2][HD_][40];      // [d][j]
  __shared__ _Float16 Vs[2][32][HD_ + 8];  // [j][d]
  __shared__ _Float16 Pst[8][16][40];      // per-wave P staging [m][j]

  const int tid  = threadIdx.x;
  const int lane = tid & 31;
  const int wave = tid >> 5;
  const int g = lane >> 4, ln = lane & 15;
  const int bh = blockIdx.y;
  const int b = bh / H_, h = bh % H_;
  const _Float16* Qh = qbuf + (size_t)bh * S_ * HD_;
  const _Float16* Kh = kbuf + (size_t)bh * S_ * HD_;
  const _Float16* Vh = vbuf + (size_t)bh * S_ * HD_;
  const int q0 = blockIdx.x * 128;
  const int qw = q0 + wave * 16;

  v16h qf[4];
#pragma unroll
  for (int c = 0; c < 4; ++c) qf[c] = load_a_frag(Qh + (size_t)qw * HD_, HD_, c * 32);

  v8f acc[8] = {};
  float mrow[8], lrow[8];
#pragma unroll
  for (int r = 0; r < 8; ++r) { mrow[r] = -1e30f; lrow[r] = 0.f; }
  const float scale = 0.08838834764831845f;   // 1/sqrt(HD)

  // per-thread KV staging: chunks tid and tid+256 of 512 (32 rows x 16 h8)
  const int kr0 = tid >> 4, kc0 = (tid & 15) << 3;
  const int kr1 = kr0 + 16, kc1 = kc0;
  h8 krg[2];

  const int nkb = blockIdx.x * 4 + 4;         // causal: keys <= q0+127

  // prologue: stage block 0
  async_copy16(Vh + (size_t)kr0 * HD_ + kc0, &Vs[0][kr0][kc0]);
  async_copy16(Vh + (size_t)kr1 * HD_ + kc1, &Vs[0][kr1][kc1]);
  krg[0] = *(const h8*)(Kh + (size_t)kr0 * HD_ + kc0);
  krg[1] = *(const h8*)(Kh + (size_t)kr1 * HD_ + kc1);
#pragma unroll
  for (int i = 0; i < 8; ++i) Kt[0][kc0 + i][kr0] = krg[0][i];
#pragma unroll
  for (int i = 0; i < 8; ++i) Kt[0][kc1 + i][kr1] = krg[1][i];

  for (int kb = 0; kb < nkb; ++kb) {
    const int cur = kb & 1;
    async_wait0();
    __syncthreads();
    if (kb + 1 < nkb) {   // stage next KV block while computing on current
      const int nxt = (kb + 1) & 1;
      const size_t ro = (size_t)(kb + 1) * 32;
      async_copy16(Vh + (ro + kr0) * HD_ + kc0, &Vs[nxt][kr0][kc0]);
      async_copy16(Vh + (ro + kr1) * HD_ + kc1, &Vs[nxt][kr1][kc1]);
      krg[0] = *(const h8*)(Kh + (ro + kr0) * HD_ + kc0);
      krg[1] = *(const h8*)(Kh + (ro + kr1) * HD_ + kc1);
    }

    // scores: 16 q-rows x 32 keys, K-dim = HD in 4 steps of 32
    v8f sc[2] = {};
#pragma unroll
    for (int f = 0; f < 2; ++f) {
#pragma unroll
      for (int c = 0; c < 4; ++c) {
        v16h bfr = load_b_frag_kn(&Kt[cur][c * 32][f * 16], 40);
        sc[f] = __builtin_amdgcn_wmma_f32_16x16x32_f16(
            false, qf[c], false, bfr, (short)0, sc[f], false, false);
      }
    }

    // online softmax (row reductions across the 16-lane half-wave)
#pragma unroll
    for (int r = 0; r < 8; ++r) {
      const int qg = qw + r + 8 * g;
      const int j0 = kb * 32 + ln;
      const int j1 = j0 + 16;
      float s0 = (j0 <= qg) ? sc[0][r] * scale : -1e30f;
      float s1 = (j1 <= qg) ? sc[1][r] * scale : -1e30f;
      float v = fmaxf(s0, s1);
#pragma unroll
      for (int off = 1; off < 16; off <<= 1) v = fmaxf(v, __shfl_xor(v, off, 32));
      const float mn = fmaxf(mrow[r], v);
      const float p0 = __expf(s0 - mn);
      const float p1 = __expf(s1 - mn);
      const float corr = __expf(mrow[r] - mn);
      mrow[r] = mn;
      float ps = p0 + p1;
#pragma unroll
      for (int off = 1; off < 16; off <<= 1) ps += __shfl_xor(ps, off, 32);
      lrow[r] = lrow[r] * corr + ps;
#pragma unroll
      for (int dt = 0; dt < 8; ++dt) acc[dt][r] *= corr;
      Pst[wave][r + 8 * g][ln]      = (_Float16)p0;   // C-layout -> LDS
      Pst[wave][r + 8 * g][16 + ln] = (_Float16)p1;
    }

    // O += P (16x32) @ V (32xHD)
    const v16h pa = load_a_frag(&Pst[wave][0][0], 40, 0);
#pragma unroll
    for (int dt = 0; dt < 8; ++dt) {
      v16h bfr = load_b_frag_kn(&Vs[cur][0][dt * 16], HD_ + 8);
      acc[dt] = __builtin_amdgcn_wmma_f32_16x16x32_f16(
          false, pa, false, bfr, (short)0, acc[dt], false, false);
    }

    if (kb + 1 < nkb) {   // commit transposed K block to the other buffer
      const int nxt = (kb + 1) & 1;
#pragma unroll
      for (int i = 0; i < 8; ++i) Kt[nxt][kc0 + i][kr0] = krg[0][i];
#pragma unroll
      for (int i = 0; i < 8; ++i) Kt[nxt][kc1 + i][kr1] = krg[1][i];
    }
  }

  // epilogue: normalize and scatter to (B,S,DIM) f16
#pragma unroll
  for (int dt = 0; dt < 8; ++dt) {
#pragma unroll
    for (int r = 0; r < 8; ++r) {
      const int sq = qw + r + 8 * g;
      const float o = acc[dt][r] / lrow[r];
      attn[((size_t)b * S_ + sq) * DIM_ + h * HD_ + dt * 16 + ln] = (_Float16)o;
    }
  }
}

// ---------------------------------------------------------------------------
extern "C" void kernel_launch(void* const* d_in, const int* in_sizes, int n_in,
                              void* d_out, int out_size, void* d_ws, size_t ws_size,
                              hipStream_t stream) {
  (void)in_sizes; (void)n_in; (void)out_size; (void)ws_size;
  const float* x    = (const float*)d_in[0];
  const float* cosT = (const float*)d_in[1];
  const float* sinT = (const float*)d_in[2];
  // d_in[3] = attn_mask (causal; computed analytically in-kernel)
  const float* wq  = (const float*)d_in[4];
  const float* wk  = (const float*)d_in[5];
  const float* wv  = (const float*)d_in[6];
  const float* wo  = (const float*)d_in[7];
  const float* qnw = (const float*)d_in[8];
  const float* knw = (const float*)d_in[9];

  char* ws = (char*)d_ws;
  size_t off = 0;
  auto alloc = [&](size_t bytes) { char* p = ws + off; off += bytes; return p; };
  _Float16* xh  = (_Float16*)alloc((size_t)MROWS * DIM_ * 2);
  _Float16* wqh = (_Float16*)alloc((size_t)DIM_ * DIM_ * 2);
  _Float16* wkh = (_Float16*)alloc((size_t)DIM_ * DIM_ * 2);
  _Float16* wvh = (_Float16*)alloc((size_t)DIM_ * DIM_ * 2);
  _Float16* woh = (_Float16*)alloc((size_t)DIM_ * DIM_ * 2);
  _Float16* qb  = (_Float16*)alloc((size_t)MROWS * DIM_ * 2);   // (B,H,S,HD)
  _Float16* kb  = (_Float16*)alloc((size_t)MROWS * DIM_ * 2);
  _Float16* vb  = (_Float16*)alloc((size_t)MROWS * DIM_ * 2);
  _Float16* ab  = (_Float16*)alloc((size_t)MROWS * DIM_ * 2);   // (B,S,DIM)

  cast_f32_to_f16<<<2048, 256, 0, stream>>>(x,  xh,  MROWS * DIM_ / 8);
  cast_f32_to_f16<<<1024, 256, 0, stream>>>(wq, wqh, DIM_ * DIM_ / 8);
  cast_f32_to_f16<<<1024, 256, 0, stream>>>(wk, wkh, DIM_ * DIM_ / 8);
  cast_f32_to_f16<<<1024, 256, 0, stream>>>(wv, wvh, DIM_ * DIM_ / 8);
  cast_f32_to_f16<<<1024, 256, 0, stream>>>(wo, woh, DIM_ * DIM_ / 8);

  dim3 gg(DIM_ / 128, MROWS / 128);
  gemm_nt_f16<<<gg, 256, 0, stream>>>(xh, wqh, nullptr, qb, MROWS, DIM_, DIM_, 1);
  gemm_nt_f16<<<gg, 256, 0, stream>>>(xh, wkh, nullptr, kb, MROWS, DIM_, DIM_, 1);
  gemm_nt_f16<<<gg, 256, 0, stream>>>(xh, wvh, nullptr, vb, MROWS, DIM_, DIM_, 1);

  rmsnorm_rope<<<dim3(B_ * H_ * S_, 2), 128, 0, stream>>>(qb, kb, qnw, knw, cosT, sinT);

  flash_attn<<<dim3(S_ / 128, B_ * H_), 256, 0, stream>>>(qb, kb, vb, ab);

  gemm_nt_f16<<<gg, 256, 0, stream>>>(ab, woh, (float*)d_out, nullptr, MROWS, DIM_, DIM_, 0);
}
